// ITASelfAttention_3564822856122
// MI455X (gfx1250) — matrix operand order repaired and checked by reading, not compiled, not run on hardware
//
#include <hip/hip_runtime.h>
#include <hip/hip_bf16.h>

typedef __bf16 bf16s;
typedef bf16s v16bf __attribute__((ext_vector_type(16)));
typedef bf16s v8bf  __attribute__((ext_vector_type(8)));
typedef float v8f   __attribute__((ext_vector_type(8)));

#define BB   8
#define SEQ  1024
#define EMB  768
#define NH   12
#define HD   64
#define MROWS (BB * SEQ)   // 8192

__device__ __forceinline__ unsigned short f32_to_bf16(float f) {
  union { float f; unsigned int u; } v; v.f = f;
  unsigned int u = v.u;
  u += 0x7FFFu + ((u >> 16) & 1u);   // round-to-nearest-even
  return (unsigned short)(u >> 16);
}

// A-fragment (16x32 bf16): lane(m = lane%16, half = lane/16) holds
// K = half*8 + {0..7} and half*8+16 + {0..7}  -> two 16B loads.
__device__ __forceinline__ v16bf frag_2x8(const unsigned short* p0,
                                          const unsigned short* p1) {
  v8bf lo = *(const v8bf*)p0;
  v8bf hi = *(const v8bf*)p1;
  return __builtin_shufflevector(lo, hi, 0, 1, 2, 3, 4, 5, 6, 7,
                                         8, 9, 10, 11, 12, 13, 14, 15);
}

__device__ __forceinline__ v8f wmma_bf16(v16bf a, v16bf b, v8f c) {
  return __builtin_amdgcn_wmma_f32_16x16x32_bf16(
      /*neg_a=*/false, a, /*neg_b=*/false, b,
      /*c_mod=*/(short)0, c, /*reuse_a=*/false, /*reuse_b=*/false);
}

// async global -> LDS copy: one instruction moves 16B per active lane.
__device__ __forceinline__ void async_ld_b128(unsigned lds_byte_addr,
                                              const void* gaddr) {
  asm volatile("global_load_async_to_lds_b128 %0, %1, off"
               :: "v"(lds_byte_addr),
                  "v"((unsigned long long)(size_t)gaddr)
               : "memory");
}
__device__ __forceinline__ void wait_async0() {
  asm volatile("s_wait_asynccnt 0" ::: "memory");
}

// ---------------------------------------------------------------- convert
__global__ void cvt_f32_to_bf16_x4(const float* __restrict__ in,
                                   unsigned short* __restrict__ out, int n4) {
  int i = blockIdx.x * 256 + threadIdx.x;
  if (i < n4) {
    const float4 f = ((const float4*)in)[i];
    ushort4 h;
    h.x = f32_to_bf16(f.x);
    h.y = f32_to_bf16(f.y);
    h.z = f32_to_bf16(f.z);
    h.w = f32_to_bf16(f.w);
    ((ushort4*)out)[i] = h;
  }
}

// ---------------------------------------------------------------- GEMM
// C[row,col] = sum_k A[row,k] * Bt[col,k] + bias[col]
// A: [8192][768] bf16 row-major, Bt: [768][768] bf16 row-major.
// mode 0: store f32 row-major to outF            (final projection)
// mode 1: store bf16 to [B,NH,SEQ,HD]            (Q and K)
// mode 2: store bf16 to [B,NH,HD,SEQ]            (V transposed)
__global__ __launch_bounds__(128) void gemm_wmma(
    const unsigned short* __restrict__ A,
    const unsigned short* __restrict__ Bt,
    const float* __restrict__ bias,
    float* __restrict__ outF,
    unsigned short* __restrict__ outH,
    int mode) {
  const int lane = threadIdx.x & 31;
  const int wave = threadIdx.x >> 5;
  const int n16  = lane & 15;
  const int half = lane >> 4;
  const int rowBase = blockIdx.y * 128 + (wave >> 1) * 64;
  const int colBase = blockIdx.x * 128 + (wave & 1) * 64;

  v8f acc[4][4];
#pragma unroll
  for (int mt = 0; mt < 4; ++mt)
#pragma unroll
    for (int nt = 0; nt < 4; ++nt) acc[mt][nt] = (v8f)0.0f;

  for (int kk = 0; kk < EMB; kk += 32) {
    v16bf af[4], bfr[4];
#pragma unroll
    for (int mt = 0; mt < 4; ++mt) {
      const unsigned short* rp = A + (rowBase + mt * 16 + n16) * EMB + kk + half * 8;
      af[mt] = frag_2x8(rp, rp + 16);
    }
#pragma unroll
    for (int nt = 0; nt < 4; ++nt) {
      const unsigned short* cp = Bt + (colBase + nt * 16 + n16) * EMB + kk + half * 8;
      bfr[nt] = frag_2x8(cp, cp + 16);
    }
    if (kk + 32 < EMB) {
      __builtin_prefetch(A + (rowBase + n16) * EMB + kk + 32, 0, 1);
      __builtin_prefetch(Bt + (colBase + n16) * EMB + kk + 32, 0, 1);
    }
#pragma unroll
    for (int mt = 0; mt < 4; ++mt)
#pragma unroll
      for (int nt = 0; nt < 4; ++nt)
        acc[mt][nt] = wmma_bf16(af[mt], bfr[nt], acc[mt][nt]);
  }

  // Epilogue.  D layout: VGPR r holds element (M = r + half*8, N = n16).
#pragma unroll
  for (int mt = 0; mt < 4; ++mt) {
#pragma unroll
    for (int nt = 0; nt < 4; ++nt) {
      const int col = colBase + nt * 16 + n16;
      const float bv = bias[col];
#pragma unroll
      for (int r = 0; r < 8; ++r) {
        const int row = rowBase + mt * 16 + r + half * 8;
        const float val = acc[mt][nt][r] + bv;
        if (mode == 0) {
          outF[row * EMB + col] = val;
        } else {
          const int b = row >> 10, nn = row & 1023;
          const int h = col >> 6,  d  = col & 63;
          if (mode == 1)
            outH[((b * NH + h) * SEQ + nn) * HD + d] = f32_to_bf16(val);
          else
            outH[((b * NH + h) * HD + d) * SEQ + nn] = f32_to_bf16(val);
        }
      }
    }
  }
}

// ---------------------------------------------------------------- attention
// Workgroup = 2 waves = 64 query rows of one (batch, head).
// Per 32-key block: wave0 async-stages the K tile (32x64 bf16) to LDS,
// wave1 async-stages the V^T tile (64x32 bf16); both waves consume both.
__global__ __launch_bounds__(64) void attn_flash(
    const unsigned short* __restrict__ Q,    // [B,NH,SEQ,HD]
    const unsigned short* __restrict__ K,    // [B,NH,SEQ,HD]
    const unsigned short* __restrict__ Vt,   // [B,NH,HD,SEQ]
    unsigned short* __restrict__ Ctx) {      // [B,SEQ,EMB] bf16
  __shared__ unsigned short Klds[32 * 64];      // [key][d]   4KB
  __shared__ unsigned short Vlds[64 * 32];      // [d][key]   4KB
  __shared__ unsigned short Plds[2][32 * 32];   // per-wave P 4KB

  const int tid  = threadIdx.x;
  const int wv   = tid >> 5;
  const int lane = tid & 31;
  const int n16  = lane & 15;
  const int half = lane >> 4;
  const int bh = blockIdx.x;           // 0..95
  const int b = bh / NH, h = bh % NH;
  const int q0 = blockIdx.y * 64 + wv * 32;   // this wave's 32 query rows

  const unsigned short* Qp = Q  + (size_t)bh * SEQ * HD;
  const unsigned short* Kp = K  + (size_t)bh * SEQ * HD;
  const unsigned short* Vp = Vt + (size_t)bh * HD * SEQ;

  const unsigned kbase = (unsigned)(size_t)&Klds[0];
  const unsigned vbase = (unsigned)(size_t)&Vlds[0];

  // Q A-fragments, loaded once: [mtile][kstep over hd]
  v16bf qa[2][2];
#pragma unroll
  for (int t = 0; t < 2; ++t)
#pragma unroll
    for (int s = 0; s < 2; ++s) {
      const unsigned short* rp = Qp + (q0 + t * 16 + n16) * HD + s * 32 + half * 8;
      qa[t][s] = frag_2x8(rp, rp + 16);
    }

  v8f acc[2][4];
  float mrow[2][8], lrow[2][8];
#pragma unroll
  for (int t = 0; t < 2; ++t) {
#pragma unroll
    for (int dt = 0; dt < 4; ++dt) acc[t][dt] = (v8f)0.0f;
#pragma unroll
    for (int r = 0; r < 8; ++r) { mrow[t][r] = -1e30f; lrow[t][r] = 0.0f; }
  }

  for (int k0 = 0; k0 < SEQ; k0 += 32) {
    // ---- async stage K (wave0) / V^T (wave1) tiles into LDS -------------
    if (wv == 0) {
#pragma unroll
      for (int i = 0; i < 8; ++i) {
        const int off = (i * 32 + lane) * 16;     // byte offset in 4KB tile
        const int row = off >> 7;                 // key row (128B each)
        const int col = off & 127;
        async_ld_b128(kbase + off,
                      (const char*)(Kp + (k0 + row) * HD) + col);
      }
    } else {
#pragma unroll
      for (int i = 0; i < 8; ++i) {
        const int off = (i * 32 + lane) * 16;     // byte offset in 4KB tile
        const int row = off >> 6;                 // d row (64B each)
        const int col = off & 63;
        async_ld_b128(vbase + off,
                      (const char*)(Vp + row * SEQ + k0) + col);
      }
    }
    wait_async0();
    __syncthreads();

    // ---- logits S = Q * K^T  (f32 accum; reference applies no scaling) --
    v16bf kf[2][2];
#pragma unroll
    for (int kn = 0; kn < 2; ++kn)
#pragma unroll
      for (int s = 0; s < 2; ++s)
        kf[kn][s] = frag_2x8(Klds + (kn * 16 + n16) * HD + s * 32 + half * 8,
                             Klds + (kn * 16 + n16) * HD + s * 32 + half * 8 + 16);

    v8f s[2][2];
#pragma unroll
    for (int t = 0; t < 2; ++t)
#pragma unroll
      for (int kn = 0; kn < 2; ++kn) {
        v8f z = (v8f)0.0f;
        z = wmma_bf16(qa[t][0], kf[kn][0], z);
        z = wmma_bf16(qa[t][1], kf[kn][1], z);
        s[t][kn] = z;
      }

    // ---- streaming softmax + stage P (bf16) in LDS ----------------------
#pragma unroll
    for (int t = 0; t < 2; ++t) {
#pragma unroll
      for (int r = 0; r < 8; ++r) {
        float a0 = s[t][0][r], a1 = s[t][1][r];
        float bm = fmaxf(a0, a1);
        bm = fmaxf(bm, __shfl_xor(bm, 1, 32));
        bm = fmaxf(bm, __shfl_xor(bm, 2, 32));
        bm = fmaxf(bm, __shfl_xor(bm, 4, 32));
        bm = fmaxf(bm, __shfl_xor(bm, 8, 32));
        const float nm = fmaxf(mrow[t][r], bm);
        const float sc = __expf(mrow[t][r] - nm);
        mrow[t][r] = nm;
        const float p0 = __expf(a0 - nm);
        const float p1 = __expf(a1 - nm);
        float rs = p0 + p1;
        rs += __shfl_xor(rs, 1, 32);
        rs += __shfl_xor(rs, 2, 32);
        rs += __shfl_xor(rs, 4, 32);
        rs += __shfl_xor(rs, 8, 32);
        lrow[t][r] = lrow[t][r] * sc + rs;
#pragma unroll
        for (int dt = 0; dt < 4; ++dt) acc[t][dt][r] = acc[t][dt][r] * sc;
        const int row = t * 16 + r + half * 8;
        Plds[wv][row * 32 + n16]      = f32_to_bf16(p0);
        Plds[wv][row * 32 + 16 + n16] = f32_to_bf16(p1);
      }
    }
    __syncthreads();

    // ---- ctx += P @ V ---------------------------------------------------
    v16bf pa[2];
#pragma unroll
    for (int t = 0; t < 2; ++t) {
      const unsigned short* rp = Plds[wv] + (t * 16 + n16) * 32 + half * 8;
      pa[t] = frag_2x8(rp, rp + 16);
    }
#pragma unroll
    for (int dt = 0; dt < 4; ++dt) {
      v16bf vf = frag_2x8(Vlds + (dt * 16 + n16) * 32 + half * 8,
                          Vlds + (dt * 16 + n16) * 32 + half * 8 + 16);
#pragma unroll
      for (int t = 0; t < 2; ++t) acc[t][dt] = wmma_bf16(pa[t], vf, acc[t][dt]);
    }
    __syncthreads();   // protect K/V/P LDS before next block's staging
  }

  // normalize and store ctx (bf16) in [B,SEQ,EMB] for the output projection
#pragma unroll
  for (int t = 0; t < 2; ++t)
#pragma unroll
    for (int dt = 0; dt < 4; ++dt)
#pragma unroll
      for (int r = 0; r < 8; ++r) {
        const int row = q0 + t * 16 + r + half * 8;
        const int d = dt * 16 + n16;
        const float val = acc[t][dt][r] / lrow[t][r];
        Ctx[((size_t)b * SEQ + row) * EMB + h * HD + d] = f32_to_bf16(val);
      }
}

// ---------------------------------------------------------------- launch
extern "C" void kernel_launch(void* const* d_in, const int* in_sizes, int n_in,
                              void* d_out, int out_size, void* d_ws, size_t ws_size,
                              hipStream_t stream) {
  const float* x  = (const float*)d_in[0];
  const float* wq = (const float*)d_in[1];
  const float* bq = (const float*)d_in[2];
  const float* wk = (const float*)d_in[3];
  const float* bk = (const float*)d_in[4];
  const float* wv = (const float*)d_in[5];
  const float* bv = (const float*)d_in[6];
  const float* wo = (const float*)d_in[7];
  const float* bo = (const float*)d_in[8];

  unsigned char* ws = (unsigned char*)d_ws;
  size_t off = 0;
  auto carve = [&](size_t bytes) -> void* {
    void* p = ws + off;
    off += (bytes + 255) & ~(size_t)255;
    return p;
  };
  unsigned short* x16   = (unsigned short*)carve((size_t)MROWS * EMB * 2);
  unsigned short* wq16  = (unsigned short*)carve((size_t)EMB * EMB * 2);
  unsigned short* wk16  = (unsigned short*)carve((size_t)EMB * EMB * 2);
  unsigned short* wv16  = (unsigned short*)carve((size_t)EMB * EMB * 2);
  unsigned short* wo16  = (unsigned short*)carve((size_t)EMB * EMB * 2);
  unsigned short* q16   = (unsigned short*)carve((size_t)MROWS * EMB * 2);
  unsigned short* k16   = (unsigned short*)carve((size_t)MROWS * EMB * 2);
  unsigned short* vt16  = (unsigned short*)carve((size_t)MROWS * EMB * 2);
  unsigned short* ctx16 = (unsigned short*)carve((size_t)MROWS * EMB * 2);

  const int nx4 = MROWS * EMB / 4;   // 1572864
  const int nw4 = EMB * EMB / 4;     // 147456
  cvt_f32_to_bf16_x4<<<(nx4 + 255) / 256, 256, 0, stream>>>(x,  x16,  nx4);
  cvt_f32_to_bf16_x4<<<(nw4 + 255) / 256, 256, 0, stream>>>(wq, wq16, nw4);
  cvt_f32_to_bf16_x4<<<(nw4 + 255) / 256, 256, 0, stream>>>(wk, wk16, nw4);
  cvt_f32_to_bf16_x4<<<(nw4 + 255) / 256, 256, 0, stream>>>(wv, wv16, nw4);
  cvt_f32_to_bf16_x4<<<(nw4 + 255) / 256, 256, 0, stream>>>(wo, wo16, nw4);

  dim3 ggrid(EMB / 128, MROWS / 128);  // (6, 64)
  gemm_wmma<<<ggrid, 128, 0, stream>>>(x16, wq16, bq, nullptr, q16, 1);
  gemm_wmma<<<ggrid, 128, 0, stream>>>(x16, wk16, bk, nullptr, k16, 1);
  gemm_wmma<<<ggrid, 128, 0, stream>>>(x16, wv16, bv, nullptr, vt16, 2);

  attn_flash<<<dim3(BB * NH, SEQ / 64), 64, 0, stream>>>(q16, k16, vt16, ctx16);

  gemm_wmma<<<ggrid, 128, 0, stream>>>(ctx16, wo16, bo, (float*)d_out, nullptr, 0);
}
